// model_24644522344577
// MI455X (gfx1250) — matrix operand order repaired and checked by reading, compiled, not run on hardware
//
#include <hip/hip_runtime.h>
#include <hip/hip_bf16.h>

typedef __bf16 bf16_t;
typedef __attribute__((ext_vector_type(16))) __bf16 v16bf;
typedef __attribute__((ext_vector_type(8)))  float  v8f;

#define Bm 16
#define Nn 512
#define Dd 256
#define Hh 8
#define DH 32
#define HID 1024
#define NBLK 16
#define BS 32
#define NKEY 160          // (2*2+1)*32
#define LEVELS 5
#define PYOUT 128
#define FLATK (LEVELS*PYOUT*Nn)   // 327680
#define ROWS (Bm*Nn)              // 8192

// ---------------------------------------------------------------- helpers
__device__ __forceinline__ v8f v8f_zero() {
  v8f z = {0.f,0.f,0.f,0.f,0.f,0.f,0.f,0.f};
  return z;
}

// A-fragment (16x32 bf16, contiguous k order): lane = M row (lane&15),
// kb = (lane>>4)*16 ; elements 0..15 -> K=kb+e.
// rowp points at A[row*lda + k0] (fp32 source, converted on the fly).
__device__ __forceinline__ v16bf load_a_frag(const float* __restrict__ rowp, int kb) {
  const float4* p0 = (const float4*)(rowp + kb);
  const float4* p1 = (const float4*)(rowp + kb + 8);
  float4 a0 = p0[0], a1 = p0[1], a2 = p1[0], a3 = p1[1];
  v16bf r;
  r[0]=(bf16_t)a0.x; r[1]=(bf16_t)a0.y; r[2]=(bf16_t)a0.z; r[3]=(bf16_t)a0.w;
  r[4]=(bf16_t)a1.x; r[5]=(bf16_t)a1.y; r[6]=(bf16_t)a1.z; r[7]=(bf16_t)a1.w;
  r[8]=(bf16_t)a2.x; r[9]=(bf16_t)a2.y; r[10]=(bf16_t)a2.z; r[11]=(bf16_t)a2.w;
  r[12]=(bf16_t)a3.x; r[13]=(bf16_t)a3.y; r[14]=(bf16_t)a3.z; r[15]=(bf16_t)a3.w;
  return r;
}

__device__ __forceinline__ int clampi(int v, int lo, int hi) {
  return v < lo ? lo : (v > hi ? hi : v);
}

// ---------------------------------------------------------------- small kernels
__global__ __launch_bounds__(256) void pos_kernel(float* __restrict__ pos) {
  int idx = blockIdx.x * 256 + threadIdx.x;      // N*D = 131072
  int n = idx >> 8, i = idx & 255;
  float ex = (float)(2 * (i / 2)) * (1.0f / 256.0f);
  float ang = (float)n * powf(10000.0f, -ex);
  pos[idx] = (i & 1) ? cosf(ang) : sinf(ang);
}

// fp32 [K x Nc] -> bf16 [Nc x K] (transposed so B-fragments are contiguous)
__global__ __launch_bounds__(256) void wconv_kernel(const float* __restrict__ W,
                                                    bf16_t* __restrict__ Wt,
                                                    int K, int Nc) {
  int idx = blockIdx.x * 256 + threadIdx.x;
  if (idx >= K * Nc) return;
  int n = idx / K, k = idx - n * K;
  Wt[idx] = (bf16_t)W[(size_t)k * Nc + n];
}

__global__ __launch_bounds__(256) void zero_kernel(float* __restrict__ p, int n) {
  int i = blockIdx.x * 256 + threadIdx.x;
  if (i < n) p[i] = 0.f;
}

// ---------------------------------------------------------------- WMMA GEMM
// C[m,n] = epilogue( A[m,:] . Wt[n,:] + bias[n] )  ; A fp32 [M x K], Wt bf16 [Nc x K]
// mode 0: C[z*cB + m*Nc + n]   mode 1 (transposed): C[z*cB + n*tstride + m]
__global__ __launch_bounds__(256)
void gemm_wmma_kernel(const float* __restrict__ A, const bf16_t* __restrict__ Wt,
                      const float* __restrict__ bias, const float* __restrict__ res,
                      float* __restrict__ C, int K, int Nc,
                      long aBatch, long cBatch, int mode, int tstride, int gelu) {
  int lane = threadIdx.x & 31;
  int wv   = threadIdx.x >> 5;
  int l15  = lane & 15;
  int hi   = lane >> 4;
  int kb   = hi * 16;
  int kbB  = hi * 16;
  long zA = (long)blockIdx.z * aBatch;
  long zC = (long)blockIdx.z * cBatch;
  int m0 = blockIdx.y * 128 + wv * 16;
  int n0 = blockIdx.x * 64;
  const float* Arow = A + zA + (size_t)(m0 + l15) * K;

  v8f acc[4];
  acc[0] = v8f_zero(); acc[1] = v8f_zero(); acc[2] = v8f_zero(); acc[3] = v8f_zero();

  for (int k0 = 0; k0 < K; k0 += 32) {
    v16bf a = load_a_frag(Arow + k0, kb);
#pragma unroll
    for (int t = 0; t < 4; t++) {
      v16bf b = *(const v16bf*)(Wt + (size_t)(n0 + t * 16 + l15) * K + k0 + kbB);
      acc[t] = __builtin_amdgcn_wmma_f32_16x16x32_bf16(
          false, a, false, b, (short)0, acc[t], false, false);
    }
  }

#pragma unroll
  for (int t = 0; t < 4; t++) {
    int n = n0 + t * 16 + l15;
    float bv = bias ? bias[n] : 0.f;
#pragma unroll
    for (int r = 0; r < 8; r++) {
      int m = m0 + r + hi * 8;
      float v = acc[t][r] + bv;
      if (gelu) {
        float x3 = v * v * v;
        v = 0.5f * v * (1.0f + tanhf(0.7978845608f * (v + 0.044715f * x3)));
      }
      if (res) v += res[(size_t)m * Nc + n];
      if (mode == 0) C[zC + (size_t)m * Nc + n] = v;
      else           C[zC + (size_t)n * tstride + m] = v;
    }
  }
}

// ---------------------------------------------------------------- LayerNorm + pos
__global__ __launch_bounds__(256)
void ln_pos_kernel(const float* __restrict__ in, float* __restrict__ out,
                   const float* __restrict__ g, const float* __restrict__ b,
                   const float* __restrict__ pos) {
  __shared__ float red[256];
  int row = blockIdx.x, t = threadIdx.x;
  float v = in[(size_t)row * Dd + t];
  red[t] = v; __syncthreads();
  for (int s = 128; s > 0; s >>= 1) { if (t < s) red[t] += red[t + s]; __syncthreads(); }
  float m = red[0] * (1.0f / Dd); __syncthreads();
  float d = v - m;
  red[t] = d * d; __syncthreads();
  for (int s = 128; s > 0; s >>= 1) { if (t < s) red[t] += red[t + s]; __syncthreads(); }
  float var = red[0] * (1.0f / Dd);
  int n = row & (Nn - 1);
  out[(size_t)row * Dd + t] = d * rsqrtf(var + 1e-5f) * g[t] + b[t] + pos[(size_t)n * Dd + t];
}

// ---------------------------------------------------------------- BatchNorm over (B,N)
__global__ __launch_bounds__(256)
void bn_stats_kernel(const float* __restrict__ x, float* __restrict__ sum,
                     float* __restrict__ sq) {
  int t = threadIdx.x;            // channel
  int r0 = blockIdx.x * 32;
  float s = 0.f, ss = 0.f;
  for (int i = 0; i < 32; i++) {
    float v = x[(size_t)(r0 + i) * Dd + t];
    s += v; ss += v * v;
  }
  atomicAdd(&sum[t], s);
  atomicAdd(&sq[t], ss);
}

__global__ __launch_bounds__(256)
void bn_apply_kernel(float* __restrict__ x, const float* __restrict__ sum,
                     const float* __restrict__ sq, const float* __restrict__ g,
                     const float* __restrict__ b) {
  int t = threadIdx.x, row = blockIdx.x;
  float m = sum[t] * (1.0f / ROWS);
  float var = sq[t] * (1.0f / ROWS) - m * m;
  float v = x[(size_t)row * Dd + t];
  x[(size_t)row * Dd + t] = (v - m) * rsqrtf(var + 1e-5f) * g[t] + b[t];
}

// ---------------------------------------------------------------- windowed attention
// one wave per (b, h, block) unit; qkv row-major [B*N x 768]; out [B*N x 256]
__global__ __launch_bounds__(64)
void attn_kernel(const float* __restrict__ qkv, float* __restrict__ out) {
  __shared__ float sP[2][32][NKEY];     // 40 KB
  int lane = threadIdx.x & 31;
  int wv   = threadIdx.x >> 5;
  int unit = blockIdx.x * 2 + wv;
  int blk = unit & 15;
  int h   = (unit >> 4) & 7;
  int b   = unit >> 7;
  int l15 = lane & 15;
  int hi  = lane >> 4;
  int kb  = hi * 16;
  int kbB = hi * 16;
  const float scale = 0.17677669529663687f;   // 1/sqrt(32)
  float (*P)[NKEY] = sP[wv];

  // --- scores = q @ k_band^T  (K = dh = 32, one WMMA step)
  for (int mt = 0; mt < 2; mt++) {
    int qtok = blk * BS + mt * 16 + l15;
    const float* qrow = qkv + (size_t)(b * Nn + qtok) * 768 + h * DH;
    v16bf aq = load_a_frag(qrow, kb);
    for (int nt = 0; nt < 10; nt++) {
      int key  = nt * 16 + l15;
      int ktok = clampi(blk * BS + key - 64, 0, Nn - 1);
      const float* krow = qkv + (size_t)(b * Nn + ktok) * 768 + 256 + h * DH + kbB;
      const float4* kp = (const float4*)krow;
      float4 k0 = kp[0], k1 = kp[1], k2 = kp[2], k3 = kp[3];
      v16bf bk;
      bk[0]=(bf16_t)k0.x; bk[1]=(bf16_t)k0.y; bk[2]=(bf16_t)k0.z; bk[3]=(bf16_t)k0.w;
      bk[4]=(bf16_t)k1.x; bk[5]=(bf16_t)k1.y; bk[6]=(bf16_t)k1.z; bk[7]=(bf16_t)k1.w;
      bk[8]=(bf16_t)k2.x; bk[9]=(bf16_t)k2.y; bk[10]=(bf16_t)k2.z; bk[11]=(bf16_t)k2.w;
      bk[12]=(bf16_t)k3.x; bk[13]=(bf16_t)k3.y; bk[14]=(bf16_t)k3.z; bk[15]=(bf16_t)k3.w;
      v8f acc = v8f_zero();
      acc = __builtin_amdgcn_wmma_f32_16x16x32_bf16(
          false, aq, false, bk, (short)0, acc, false, false);
      int rel = (key >> 5) - 2;
      float mask = ((unsigned)(blk + rel) < (unsigned)NBLK) ? 1.f : 0.f;
#pragma unroll
      for (int r = 0; r < 8; r++) {
        int rowi = mt * 16 + r + hi * 8;
        P[rowi][key] = mask * acc[r] * scale + (1.f - mask) * -1e9f;
      }
    }
  }
  __syncthreads();

  // --- softmax over 160 keys; lane = row
  {
    float mx = -3e38f;
    for (int j = 0; j < NKEY; j++) mx = fmaxf(mx, P[lane][j]);
    float s = 0.f;
    for (int j = 0; j < NKEY; j++) { float e = __expf(P[lane][j] - mx); P[lane][j] = e; s += e; }
    float inv = 1.0f / s;
    for (int j = 0; j < NKEY; j++) P[lane][j] *= inv;
  }
  __syncthreads();

  // --- o = P @ v_band   (K = 160 -> 5 WMMA steps)
  for (int mt = 0; mt < 2; mt++) {
    v8f acc[2]; acc[0] = v8f_zero(); acc[1] = v8f_zero();
    for (int ks = 0; ks < 5; ks++) {
      const float* prow = &P[mt * 16 + l15][ks * 32 + kb];
      v16bf ap;
#pragma unroll
      for (int e = 0; e < 8; e++) ap[e] = (bf16_t)prow[e];
#pragma unroll
      for (int e = 0; e < 8; e++) ap[8 + e] = (bf16_t)prow[8 + e];
#pragma unroll
      for (int nt = 0; nt < 2; nt++) {
        int col = nt * 16 + l15;
        v16bf bv;
#pragma unroll
        for (int e = 0; e < 16; e++) {
          int keyi = ks * 32 + kbB + e;
          int vtok = clampi(blk * BS + keyi - 64, 0, Nn - 1);
          bv[e] = (bf16_t)qkv[(size_t)(b * Nn + vtok) * 768 + 512 + h * DH + col];
        }
        acc[nt] = __builtin_amdgcn_wmma_f32_16x16x32_bf16(
            false, ap, false, bv, (short)0, acc[nt], false, false);
      }
    }
#pragma unroll
    for (int nt = 0; nt < 2; nt++) {
#pragma unroll
      for (int r = 0; r < 8; r++) {
        int tok = blk * BS + mt * 16 + r + hi * 8;
        out[(size_t)(b * Nn + tok) * Dd + h * DH + nt * 16 + l15] = acc[nt][r];
      }
    }
  }
}

// ---------------------------------------------------------------- fcl: [16 x 327680] @ [327680 x 128]
// bandwidth-bound on fcl_W (168 MB); split-K with coalesced W reads + atomics
__global__ __launch_bounds__(128)
void fcl_kernel(const float* __restrict__ feats, const float* __restrict__ W,
                float* __restrict__ add) {
  int t = threadIdx.x;                   // output channel j
  int k0 = blockIdx.x * 640;
  float acc[Bm];
#pragma unroll
  for (int i = 0; i < Bm; i++) acc[i] = 0.f;
  for (int k = k0; k < k0 + 640; k++) {
    float wv = W[(size_t)k * 128 + t];
#pragma unroll
    for (int bb = 0; bb < Bm; bb++) acc[bb] += feats[(size_t)bb * FLATK + k] * wv;
  }
#pragma unroll
  for (int bb = 0; bb < Bm; bb++) atomicAdd(&add[bb * 128 + t], acc[bb]);
}

// add bias, BatchNorm over batch (training-mode stats over B=16)
__global__ __launch_bounds__(128)
void addbn_kernel(const float* __restrict__ add, const float* __restrict__ fcl_b,
                  const float* __restrict__ g, const float* __restrict__ bta,
                  float* __restrict__ outbn) {
  int c = threadIdx.x;
  float vals[Bm]; float s = 0.f;
  for (int bb = 0; bb < Bm; bb++) { float v = add[bb * 128 + c] + fcl_b[c]; vals[bb] = v; s += v; }
  float m = s * (1.0f / Bm); float ss = 0.f;
  for (int bb = 0; bb < Bm; bb++) { float d = vals[bb] - m; ss += d * d; }
  float inv = rsqrtf(ss * (1.0f / Bm) + 1e-5f);
  for (int bb = 0; bb < Bm; bb++) outbn[bb * 128 + c] = (vals[bb] - m) * inv * g[c] + bta[c];
}

// detection heads + softmax + pack output [16, 9, 9]
__global__ __launch_bounds__(256)
void heads_kernel(const float* __restrict__ addbn,
                  const float* __restrict__ cW1, const float* __restrict__ cb1,
                  const float* __restrict__ cW2, const float* __restrict__ cb2,
                  const float* __restrict__ bW1, const float* __restrict__ bb1,
                  const float* __restrict__ bW2, const float* __restrict__ bb2,
                  float* __restrict__ out) {
  __shared__ float h1[256], hb[128], cls[45], box[36];
  int b = blockIdx.x, t = threadIdx.x;
  const float* a = addbn + b * 128;
  { float s = cb1[t]; for (int c = 0; c < 128; c++) s += a[c] * cW1[c * 256 + t]; h1[t] = fmaxf(s, 0.f); }
  if (t < 128) { float s = bb1[t]; for (int c = 0; c < 128; c++) s += a[c] * bW1[c * 128 + t]; hb[t] = fmaxf(s, 0.f); }
  __syncthreads();
  if (t < 45) { float s = cb2[t]; for (int j = 0; j < 256; j++) s += h1[j] * cW2[j * 45 + t]; cls[t] = s; }
  if (t >= 64 && t < 100) { int u = t - 64; float s = bb2[u]; for (int j = 0; j < 128; j++) s += hb[j] * bW2[j * 36 + u]; box[u] = s; }
  __syncthreads();
  if (t == 0) {
    float mx = -3e38f; for (int j = 0; j < 45; j++) mx = fmaxf(mx, cls[j]);
    float s = 0.f; for (int j = 0; j < 45; j++) { cls[j] = __expf(cls[j] - mx); s += cls[j]; }
    float inv = 1.f / s; for (int j = 0; j < 45; j++) cls[j] *= inv;
  }
  __syncthreads();
  if (t < 81) {
    int a9 = t / 9, k = t - a9 * 9;
    out[b * 81 + t] = (k < 5) ? cls[a9 * 5 + k] : box[a9 * 4 + (k - 5)];
  }
}

// ---------------------------------------------------------------- host orchestration
extern "C" void kernel_launch(void* const* d_in, const int* in_sizes, int n_in,
                              void* d_out, int out_size, void* d_ws, size_t ws_size,
                              hipStream_t stream) {
  const float* x       = (const float*)d_in[0];
  const float* patch_W = (const float*)d_in[1];
  const float* patch_b = (const float*)d_in[2];
  const float* ln_g    = (const float*)d_in[3];
  const float* ln_b    = (const float*)d_in[4];
  const float* ff_W1   = (const float*)d_in[5];
  const float* ff_b1   = (const float*)d_in[6];
  const float* ff_W2   = (const float*)d_in[7];
  const float* ff_b2   = (const float*)d_in[8];
  const float* Wqkv    = (const float*)d_in[9];
  const float* bqkv    = (const float*)d_in[10];
  const float* Wo      = (const float*)d_in[11];
  const float* bo      = (const float*)d_in[12];
  const float* bn1_g   = (const float*)d_in[13];
  const float* bn1_b   = (const float*)d_in[14];
  const float* bn2_g   = (const float*)d_in[15];
  const float* bn2_b   = (const float*)d_in[16];
  const float* pyr_W   = (const float*)d_in[17];
  const float* pyr_b   = (const float*)d_in[18];
  const float* fcl_W   = (const float*)d_in[19];
  const float* fcl_b   = (const float*)d_in[20];
  const float* addbn_g = (const float*)d_in[21];
  const float* addbn_b = (const float*)d_in[22];
  const float* cls_W1  = (const float*)d_in[23];
  const float* cls_b1  = (const float*)d_in[24];
  const float* cls_W2  = (const float*)d_in[25];
  const float* cls_b2  = (const float*)d_in[26];
  const float* box_W1  = (const float*)d_in[27];
  const float* box_b1  = (const float*)d_in[28];
  const float* box_W2  = (const float*)d_in[29];
  const float* box_b2  = (const float*)d_in[30];
  float* out = (float*)d_out;

  char* wsp = (char*)d_ws;
  auto carve = [&](size_t bytes) -> void* {
    void* p = (void*)wsp;
    wsp += (bytes + 255) & ~(size_t)255;
    return p;
  };
  float* POS   = (float*)carve((size_t)Nn * Dd * 4);
  float* X0    = (float*)carve((size_t)ROWS * Dd * 4);
  float* T0    = (float*)carve((size_t)ROWS * Dd * 4);
  float* H1    = (float*)carve((size_t)ROWS * Dd * 4);
  float* FFMID = (float*)carve((size_t)ROWS * HID * 4);
  float* QKV   = (float*)carve((size_t)ROWS * 768 * 4);
  float* FEATS = (float*)carve((size_t)Bm * FLATK * 4);
  float* ADD   = (float*)carve(Bm * 128 * 4);
  float* ADDBN = (float*)carve(Bm * 128 * 4);
  float* BNSUM = (float*)carve(Dd * 4);
  float* BNSQ  = (float*)carve(Dd * 4);
  bf16_t* patchWt = (bf16_t*)carve((size_t)Dd * Dd * 2);
  bf16_t* ffW1t   = (bf16_t*)carve((size_t)Dd * HID * 2);
  bf16_t* ffW2t   = (bf16_t*)carve((size_t)HID * Dd * 2);
  bf16_t* qkvWt   = (bf16_t*)carve((size_t)Dd * 768 * 2);
  bf16_t* woWt    = (bf16_t*)carve((size_t)Dd * Dd * 2);
  bf16_t* pyrWt   = (bf16_t*)carve((size_t)LEVELS * Dd * PYOUT * 2);

  // positional table + weight conversion (each call: deterministic, no caching)
  pos_kernel<<<Nn * Dd / 256, 256, 0, stream>>>(POS);
  wconv_kernel<<<(Dd * Dd + 255) / 256, 256, 0, stream>>>(patch_W, patchWt, Dd, Dd);
  wconv_kernel<<<(Dd * HID + 255) / 256, 256, 0, stream>>>(ff_W1, ffW1t, Dd, HID);
  wconv_kernel<<<(HID * Dd + 255) / 256, 256, 0, stream>>>(ff_W2, ffW2t, HID, Dd);
  wconv_kernel<<<(Dd * 768 + 255) / 256, 256, 0, stream>>>(Wqkv, qkvWt, Dd, 768);
  wconv_kernel<<<(Dd * Dd + 255) / 256, 256, 0, stream>>>(Wo, woWt, Dd, Dd);
  for (int i = 0; i < LEVELS; i++)
    wconv_kernel<<<(Dd * PYOUT + 255) / 256, 256, 0, stream>>>(
        pyr_W + (size_t)i * Dd * PYOUT, pyrWt + (size_t)i * PYOUT * Dd, Dd, PYOUT);

  auto gemm = [&](const float* A, const bf16_t* Wt, const float* bias, const float* res,
                  float* C, int K, int Nc, int gelu) {
    dim3 g(Nc / 64, ROWS / 128, 1);
    gemm_wmma_kernel<<<g, 256, 0, stream>>>(A, Wt, bias, res, C, K, Nc, 0, 0, 0, 0, gelu);
  };
  auto bn = [&](float* buf, const float* g, const float* b) {
    zero_kernel<<<2, 256, 0, stream>>>(BNSUM, Dd);
    zero_kernel<<<2, 256, 0, stream>>>(BNSQ, Dd);
    bn_stats_kernel<<<ROWS / 32, 256, 0, stream>>>(buf, BNSUM, BNSQ);
    bn_apply_kernel<<<ROWS, 256, 0, stream>>>(buf, BNSUM, BNSQ, g, b);
  };

  const float* xcur = x;
  for (int it = 0; it < 5; it++) {
    gemm(xcur, patchWt, patch_b, nullptr, T0, Dd, Dd, 0);          // patch embed
    ln_pos_kernel<<<ROWS, 256, 0, stream>>>(T0, H1, ln_g, ln_b, POS);
    gemm(H1, ffW1t, ff_b1, nullptr, FFMID, Dd, HID, 1);            // FFN up + GELU
    gemm(FFMID, ffW2t, ff_b2, nullptr, T0, HID, Dd, 0);            // FFN down
    gemm(T0, qkvWt, bqkv, nullptr, QKV, Dd, 768, 0);               // QKV
    attn_kernel<<<(Bm * Hh * NBLK) / 2, 64, 0, stream>>>(QKV, FFMID); // attention (reuse FFMID)
    gemm(FFMID, woWt, bo, H1, T0, Dd, Dd, 0);                      // Wo + residual(idt)
    bn(T0, bn1_g, bn1_b);
    gemm(T0, ffW1t, ff_b1, nullptr, FFMID, Dd, HID, 1);            // FFN up + GELU
    gemm(FFMID, ffW2t, ff_b2, T0, X0, HID, Dd, 0);                 // FFN down + residual
    bn(X0, bn2_g, bn2_b);
    // pyramid level `it` into feats (transposed store: feats[b, it, c, n])
    {
      dim3 g(PYOUT / 64, Nn / 128, Bm);
      gemm_wmma_kernel<<<g, 256, 0, stream>>>(
          X0, pyrWt + (size_t)it * PYOUT * Dd, pyr_b + it * PYOUT, nullptr,
          FEATS + (size_t)it * PYOUT * Nn, Dd, PYOUT,
          (long)Nn * Dd, (long)FLATK, 1, Nn, 0);
    }
    xcur = X0;
  }

  zero_kernel<<<8, 256, 0, stream>>>(ADD, Bm * 128);
  fcl_kernel<<<FLATK / 640, 128, 0, stream>>>(FEATS, fcl_W, ADD);
  addbn_kernel<<<1, 128, 0, stream>>>(ADD, fcl_b, addbn_g, addbn_b, ADDBN);
  heads_kernel<<<Bm, 256, 0, stream>>>(ADDBN, cls_W1, cls_b1, cls_W2, cls_b2,
                                       box_W1, box_b1, box_W2, box_b2, out);
  (void)in_sizes; (void)n_in; (void)out_size; (void)ws_size;
}
